// SS2D_1_40664750358805
// MI455X (gfx1250) — matrix operand order repaired and checked by reading, not compiled, not run on hardware
//
#include <hip/hip_runtime.h>
#include <hip/hip_bf16.h>

// ---------------------------------------------------------------------------
// SS2D forward for MI455X (gfx1250, wave32).
//   B=2, D_MODEL=96, D_EXP=192, H=W=48, L=2304, D_STATE=16, DT_RANK=6
// GEMMs build WMMA fragments DIRECTLY from global memory (no LDS staging):
//   A fragment: lane = M row, K-pairs contiguous  -> 8x float2 per lane
//   B fragment: lane = N column, 16 K values      -> 16 scalar loads per lane
// Weight fragments are preloaded in VGPRs and reused across many tiles.
// Scan uses gfx1250 async global->LDS double-buffering when available.
// ---------------------------------------------------------------------------

typedef _Float16 f16;
typedef __attribute__((ext_vector_type(16))) _Float16 v16h;
typedef __attribute__((ext_vector_type(8)))  float    v8f;

#define D_MODEL 96
#define D_EXP   192
#define NPIX    2304   // 48*48
#define NPROJ   152    // 4*(6+16+16)
#define DSTATE  16

#if __has_builtin(__builtin_amdgcn_global_load_async_to_lds_b32) && \
    __has_builtin(__builtin_amdgcn_s_wait_asynccnt)
#define HAVE_ASYNC_LDS 1
typedef __attribute__((address_space(1))) int as1_int;   // global AS pointer elem
typedef __attribute__((address_space(3))) int as3_int;   // LDS AS pointer elem
#else
#define HAVE_ASYNC_LDS 0
#endif

__device__ __forceinline__ float sigmoidf_(float x) {
    return __builtin_amdgcn_rcpf(1.0f + __expf(-x));   // v_rcp_f32, no IEEE div
}
__device__ __forceinline__ float siluf_(float x)    { return x * sigmoidf_(x); }
__device__ __forceinline__ float softplusf_(float x){
    return (x > 20.0f) ? x : __logf(1.0f + __expf(x)); // v_log_f32 fast path
}

// pixel index visited by direction k at scan step s
__device__ __forceinline__ int pixel_of(int k, int s) {
    switch (k & 3) {
        case 0:  return s;
        case 1:  return (s % 48) * 48 + (s / 48);
        case 2:  return 2303 - s;
        default: { int t = 2303 - s; return (t % 48) * 48 + (t / 48); }
    }
}

__device__ __forceinline__ v8f wmma_(v16h a, v16h b, v8f c) {
    return __builtin_amdgcn_wmma_f32_16x16x32_f16(false, a, false, b,
                                                  (short)0, c, false, false);
}

// A fragment from a row-major-in-K source. `base` already points at
// (row = lane&15, k0). ISA 7.12.2: VGPR j<4 -> K = half*8+2j, j>=4 -> +16.
__device__ __forceinline__ v16h frag_a_rowmajor(const float* __restrict__ base, int lane) {
    v16h a;
    const int half = lane >> 4;
#pragma unroll
    for (int j = 0; j < 8; ++j) {
        const int kk = (j < 4) ? (half * 8 + 2 * j) : (16 + half * 8 + 2 * (j - 4));
        const float2 p = *(const float2*)(base + kk);
        a[2 * j]     = (f16)p.x;
        a[2 * j + 1] = (f16)p.y;
    }
    return a;
}

// B fragment from a K-strided (stride NPIX) source. `p` points at (k0, N = lane&15).
__device__ __forceinline__ v16h frag_b_contig(const float* __restrict__ p, int lane) {
    v16h b;
    const int half = lane >> 4;
#pragma unroll
    for (int s = 0; s < 16; ++s) {
        const int kk = (s < 8) ? (half * 8 + s) : (16 + half * 8 + (s - 8));
        b[s] = (f16)p[(size_t)kk * NPIX];
    }
    return b;
}

// ============================ K1: in_proj GEMM =============================
// Orientation: M = 384 output channels, N = tokens.
// A = in_w (384x96 row-major), preloaded; B = x (channel-major, token-contig).
// xz[b, ch, p] = W_in[ch,:].x[b,:,p] + b_in[ch];  ch>=192 -> silu (z half)
__global__ __launch_bounds__(32) void k_in_proj(const float* __restrict__ x,
                                                const float* __restrict__ W,
                                                const float* __restrict__ bias,
                                                float* __restrict__ xz) {
    const int lane = threadIdx.x;
    const int m0 = blockIdx.x * 16;      // channel tile
    const int tchunk = blockIdx.y;       // 16 token tiles per block
    const int b = blockIdx.z;
    const float* xb = x + (size_t)b * D_MODEL * NPIX;
    const bool dosilu = (m0 >= D_EXP);   // tile-uniform: z half

    v16h afrag[3];
    const float* wrow = W + (size_t)(m0 + (lane & 15)) * D_MODEL;
#pragma unroll
    for (int ks = 0; ks < 3; ++ks) afrag[ks] = frag_a_rowmajor(wrow + ks * 32, lane);

    float bv[8];
#pragma unroll
    for (int r = 0; r < 8; ++r) bv[r] = bias[m0 + r + (lane >> 4) * 8];

    for (int tt = 0; tt < 16; ++tt) {
        const int n0 = (tchunk * 16 + tt) * 16;          // token base
        v8f acc = {};
#pragma unroll
        for (int ks = 0; ks < 3; ++ks)
            acc = wmma_(afrag[ks], frag_b_contig(xb + (size_t)(ks * 32) * NPIX + n0 + (lane & 15), lane), acc);
        const int tok = n0 + (lane & 15);
#pragma unroll
        for (int r = 0; r < 8; ++r) {
            const int ch = m0 + r + (lane >> 4) * 8;
            float v = acc[r] + bv[r];
            if (dosilu) v = siluf_(v);                   // branchless-cheap silu
            xz[(size_t)b * 2 * D_EXP * NPIX + (size_t)ch * NPIX + tok] = v;
        }
    }
}

// ======================= K2: depthwise 3x3 + silu ==========================
__global__ __launch_bounds__(256) void k_dwconv(const float* __restrict__ xz,
                                                const float* __restrict__ dww,
                                                const float* __restrict__ dwb,
                                                float* __restrict__ x1s) {
    const int idx = blockIdx.x * 256 + threadIdx.x;     // 2*2304*192 total
    const int d = idx % D_EXP;
    const int t = idx / D_EXP;
    const int p = t % NPIX, b = t / NPIX;
    const int yy = p / 48, xx = p % 48;
    const float* src = xz + (size_t)b * 2 * D_EXP * NPIX + (size_t)d * NPIX;
    const float* wv = dww + d * 9;
    float acc = dwb[d];
#pragma unroll
    for (int dy = -1; dy <= 1; ++dy)
#pragma unroll
        for (int dx = -1; dx <= 1; ++dx) {
            const int Y = yy + dy, X = xx + dx;
            if (Y >= 0 && Y < 48 && X >= 0 && X < 48)
                acc += src[Y * 48 + X] * wv[(dy + 1) * 3 + (dx + 1)];
        }
    x1s[(size_t)t * D_EXP + d] = siluf_(acc);   // token-major
}

// =================== K3: x_proj GEMM (all 4 directions) ====================
// M = tokens, N = 152 proj channels (NT=2 accumulators, B preloaded).
// proj[b, p, k*38+c] = x_proj_w[k,c,:] . x1s[b,p,:]
__global__ __launch_bounds__(32) void k_proj(const float* __restrict__ x1s,
                                             const float* __restrict__ xpw,
                                             float* __restrict__ proj) {
    const int lane = threadIdx.x;
    const int mchunk = blockIdx.x;       // 3 token tiles per block
    const int npair = blockIdx.y;        // 32-wide N slab
    const int b = blockIdx.z;

    v16h bfrag[6][2];
#pragma unroll
    for (int ks = 0; ks < 6; ++ks)
#pragma unroll
        for (int nt = 0; nt < 2; ++nt) {
            v16h bb;
#pragma unroll
            for (int e = 0; e < 16; ++e) {
                const int n = npair * 32 + nt * 16 + (lane & 15);
                const int kk = (e < 8) ? ((lane >> 4) * 8 + e) : (16 + (lane >> 4) * 8 + (e - 8));
                float w = 0.0f;
                if (n < NPROJ)
                    w = xpw[(size_t)(n / 38) * 38 * D_EXP + (size_t)(n % 38) * D_EXP
                            + ks * 32 + kk];
                bb[e] = (f16)w;
            }
            bfrag[ks][nt] = bb;
        }

    const float* xbase = x1s + (size_t)b * NPIX * D_EXP;
    for (int mt = 0; mt < 3; ++mt) {
        const int m0 = (mchunk * 3 + mt) * 16;
        const float* arow = xbase + (size_t)(m0 + (lane & 15)) * D_EXP;
        v8f acc[2] = {{}, {}};
#pragma unroll
        for (int ks = 0; ks < 6; ++ks) {
            const v16h a = frag_a_rowmajor(arow + ks * 32, lane);
#pragma unroll
            for (int nt = 0; nt < 2; ++nt)
                acc[nt] = wmma_(a, bfrag[ks][nt], acc[nt]);
        }
#pragma unroll
        for (int nt = 0; nt < 2; ++nt) {
            const int n = npair * 32 + nt * 16 + (lane & 15);
            if (n < NPROJ) {
#pragma unroll
                for (int r = 0; r < 8; ++r) {
                    const int m = m0 + r + (lane >> 4) * 8;
                    proj[((size_t)b * NPIX + m) * NPROJ + n] = acc[nt][r];
                }
            }
        }
    }
}

// ====================== K4: 4-direction selective scan =====================
// 48 one-wave blocks: (b,k,dblock). lane = d within block of 32.
// State h[16] in registers. The per-step 38-float proj row is shared across
// lanes: double-buffer it into LDS with gfx1250 async global->LDS DMA
// (ASYNCcnt) when the toolchain exposes the builtin; else uniform s_loads.
__global__ __launch_bounds__(32) void k_scan(const float* __restrict__ x1s,
                                             const float* __restrict__ proj,
                                             const float* __restrict__ dt_w,
                                             const float* __restrict__ dt_b,
                                             const float* __restrict__ A_logs,
                                             const float* __restrict__ Ds,
                                             float* __restrict__ y4) {
    const int blk = blockIdx.x;
    const int b = blk / 24, k = (blk / 6) % 4, dblk = blk % 6;
    const int lane = threadIdx.x;
    const int d = dblk * 32 + lane;
    const int kd = k * D_EXP + d;

    float Arow[DSTATE];
#pragma unroll
    for (int n = 0; n < DSTATE; ++n) Arow[n] = -__expf(A_logs[(size_t)kd * DSTATE + n]);
    float dtw[6];
#pragma unroll
    for (int j = 0; j < 6; ++j) dtw[j] = dt_w[(size_t)kd * 6 + j];
    const float dtb = dt_b[kd];
    const float Dv = Ds[kd];
    float h[DSTATE];
#pragma unroll
    for (int n = 0; n < DSTATE; ++n) h[n] = 0.0f;

    const float* projb = proj + (size_t)b * NPIX * NPROJ + k * 38;
    const float* ub = x1s + (size_t)b * NPIX * D_EXP + d;
    float* yb = y4 + ((size_t)(b * 4 + k)) * NPIX * D_EXP + d;

#if HAVE_ASYNC_LDS
    __shared__ float pbuf[2][64];
    // issue one proj row (38 floats, padded) as 2 lane-wide async b32 ops
    auto issue_row = [&](int s, int bufi) {
        const int p = pixel_of(k, s);
        const float* src = projb + (size_t)p * NPROJ + lane;
        float* dst = &pbuf[bufi][lane];
        __builtin_amdgcn_global_load_async_to_lds_b32(
            (as1_int*)(void*)src, (as3_int*)(void*)dst, 0, 0);
        __builtin_amdgcn_global_load_async_to_lds_b32(
            (as1_int*)(void*)(src + 32), (as3_int*)(void*)(dst + 32), 0, 0);
    };
    issue_row(0, 0);
#endif

    for (int s = 0; s < NPIX; ++s) {
        const int p = pixel_of(k, s);
#if HAVE_ASYNC_LDS
        const int bufi = s & 1;
        const int snext = (s + 1 < NPIX) ? (s + 1) : s;
        issue_row(snext, bufi ^ 1);
        __builtin_amdgcn_s_wait_asynccnt(2);   // oldest pair (current buf) done
        const float* pp = &pbuf[bufi][0];
#else
        if (s + 16 < NPIX) {  // gfx1250 global_prefetch_b8 on upcoming proj row
            const int pn = pixel_of(k, s + 16);
            __builtin_prefetch(projb + (size_t)pn * NPROJ, 0, 1);
        }
        const float* pp = projb + (size_t)p * NPROJ;   // uniform across lanes
#endif
        const float u = ub[(size_t)p * D_EXP];
        float dt = dtb;
#pragma unroll
        for (int j = 0; j < 6; ++j) dt += dtw[j] * pp[j];
        dt = softplusf_(dt);
        const float du = dt * u;
        float y = Dv * u;
#pragma unroll
        for (int n = 0; n < DSTATE; ++n) {
            const float dA = __expf(dt * Arow[n]);
            h[n] = dA * h[n] + du * pp[6 + n];
            y += h[n] * pp[22 + n];
        }
        yb[(size_t)p * D_EXP] = y;   // lands at its pixel; flips handled by p
    }
}

// ================= K5: combine 4 directions + LayerNorm ====================
__global__ __launch_bounds__(32) void k_combine_ln(const float* __restrict__ y4,
                                                   const float* __restrict__ onw,
                                                   const float* __restrict__ onb,
                                                   float* __restrict__ yln) {
    const int b = blockIdx.x / NPIX, p = blockIdx.x % NPIX;
    const int lane = threadIdx.x;
    float v[6];
    float s = 0.0f;
#pragma unroll
    for (int j = 0; j < 6; ++j) {
        const int d = lane + 32 * j;
        float acc = 0.0f;
#pragma unroll
        for (int k = 0; k < 4; ++k)
            acc += y4[(((size_t)(b * 4 + k)) * NPIX + p) * D_EXP + d];
        v[j] = acc;
        s += acc;
    }
#pragma unroll
    for (int off = 16; off > 0; off >>= 1) s += __shfl_xor(s, off, 32);
    const float mu = s * (1.0f / D_EXP);
    float vs = 0.0f;
#pragma unroll
    for (int j = 0; j < 6; ++j) { const float dd = v[j] - mu; vs += dd * dd; }
#pragma unroll
    for (int off = 16; off > 0; off >>= 1) vs += __shfl_xor(vs, off, 32);
    const float inv = rsqrtf(vs * (1.0f / D_EXP) + 1e-5f);
#pragma unroll
    for (int j = 0; j < 6; ++j) {
        const int d = lane + 32 * j;
        yln[((size_t)b * NPIX + p) * D_EXP + d] = (v[j] - mu) * inv * onw[d] + onb[d];
    }
}

// ========================= K6: channel branch ==============================
__global__ __launch_bounds__(256) void k_channel(const float* __restrict__ yln,
                                                 const float* __restrict__ cin_w,
                                                 const float* __restrict__ cin_b,
                                                 const float* __restrict__ cout_w,
                                                 const float* __restrict__ cout_b,
                                                 const float* __restrict__ xcpw,
                                                 const float* __restrict__ dtcw,
                                                 const float* __restrict__ dtcb,
                                                 const float* __restrict__ Ac_logs,
                                                 const float* __restrict__ Dsc,
                                                 const float* __restrict__ cnw,
                                                 const float* __restrict__ cnb,
                                                 float* __restrict__ cbuf) {
    __shared__ float xc[D_EXP];
    __shared__ float ycacc[2][D_EXP];
    __shared__ float rbuf[256];
    const int b = blockIdx.x, t = threadIdx.x;

    if (t < D_EXP) {  // spatial mean per channel
        float s = 0.0f;
        const float* base = yln + (size_t)b * NPIX * D_EXP + t;
        for (int p = 0; p < NPIX; ++p) s += base[(size_t)p * D_EXP];
        xc[t] = s * (1.0f / NPIX);
    }
    __syncthreads();

    if (t < 2) {  // two direction scans over the 192 channels
        const int kc = t;
        float h[4][DSTATE];
        float Ac[4][DSTATE];
#pragma unroll
        for (int dci = 0; dci < 4; ++dci)
#pragma unroll
            for (int n = 0; n < DSTATE; ++n) {
                h[dci][n] = 0.0f;
                Ac[dci][n] = -__expf(Ac_logs[(kc * 4 + dci) * DSTATE + n]);
            }
        for (int s = 0; s < D_EXP; ++s) {
            const int pos = (kc == 0) ? s : (D_EXP - 1 - s);
            float xv[4];
#pragma unroll
            for (int dci = 0; dci < 4; ++dci) xv[dci] = xc[pos] * cin_w[dci] + cin_b[dci];
            float prj[38];
#pragma unroll
            for (int c = 0; c < 38; ++c) {
                float a = 0.0f;
#pragma unroll
                for (int dci = 0; dci < 4; ++dci) a += xcpw[kc * 38 * 4 + c * 4 + dci] * xv[dci];
                prj[c] = a;
            }
            float yout = 0.0f;
#pragma unroll
            for (int dci = 0; dci < 4; ++dci) {
                float dt = dtcb[kc * 4 + dci];
#pragma unroll
                for (int j = 0; j < 6; ++j) dt += dtcw[(kc * 4 + dci) * 6 + j] * prj[j];
                dt = softplusf_(dt);
                const float du = dt * xv[dci];
                float yd = Dsc[kc * 4 + dci] * xv[dci];
#pragma unroll
                for (int n = 0; n < DSTATE; ++n) {
                    const float dA = __expf(dt * Ac[dci][n]);
                    h[dci][n] = dA * h[dci][n] + du * prj[6 + n];
                    yd += h[dci][n] * prj[22 + n];
                }
                yout += yd * cout_w[dci];
            }
            ycacc[kc][pos] = yout;
        }
    }
    __syncthreads();

    // LayerNorm over the 192 positions (which index D_EXP channels)
    const float val = (t < D_EXP) ? (ycacc[0][t] + ycacc[1][t] + cout_b[0]) : 0.0f;
    rbuf[t] = val;
    __syncthreads();
    for (int off = 128; off > 0; off >>= 1) { if (t < off) rbuf[t] += rbuf[t + off]; __syncthreads(); }
    const float mu = rbuf[0] * (1.0f / D_EXP);
    __syncthreads();
    const float dv = (t < D_EXP) ? (val - mu) : 0.0f;
    rbuf[t] = dv * dv;
    __syncthreads();
    for (int off = 128; off > 0; off >>= 1) { if (t < off) rbuf[t] += rbuf[t + off]; __syncthreads(); }
    const float inv = rsqrtf(rbuf[0] * (1.0f / D_EXP) + 1e-5f);
    if (t < D_EXP)
        cbuf[b * D_EXP + t] = (val - mu) * inv * cnw[t] + cnb[t];
}

// ===================== K7: out_proj GEMM with fused gate ===================
// Orientation: M = 96 output channels (A = out_w preloaded), N = tokens.
// out[b,ch,p] = out_w[ch,:] . (yln[b,p,:] * c[b,:] * z[b,:,p]) + out_b[ch]
__global__ __launch_bounds__(32) void k_out_proj(const float* __restrict__ yln,
                                                 const float* __restrict__ cbuf,
                                                 const float* __restrict__ xz,
                                                 const float* __restrict__ W,
                                                 const float* __restrict__ bias,
                                                 float* __restrict__ out) {
    const int lane = threadIdx.x;
    const int m0 = blockIdx.x * 16;      // output channel tile (0..95)
    const int tchunk = blockIdx.y;       // 9 token tiles per block
    const int b = blockIdx.z;

    v16h afrag[6];
    const float* wrow = W + (size_t)(m0 + (lane & 15)) * D_EXP;
#pragma unroll
    for (int ks = 0; ks < 6; ++ks) afrag[ks] = frag_a_rowmajor(wrow + ks * 32, lane);

    float bv[8];
#pragma unroll
    for (int r = 0; r < 8; ++r) bv[r] = bias[m0 + r + (lane >> 4) * 8];

    const float* yb = yln + (size_t)b * NPIX * D_EXP;
    const float* zb = xz + (size_t)b * 2 * D_EXP * NPIX + (size_t)D_EXP * NPIX; // silu'd z
    const float* cb = cbuf + b * D_EXP;

    for (int tt = 0; tt < 9; ++tt) {
        const int n0 = (tchunk * 9 + tt) * 16;
        v8f acc = {};
#pragma unroll
        for (int ks = 0; ks < 6; ++ks) {
            const int tn = n0 + (lane & 15);            // N column for this lane
            v16h bf;
#pragma unroll
            for (int e = 0; e < 16; ++e) {
                const int d = ks * 32 + ((e < 8) ? ((lane >> 4) * 8 + e) : (16 + (lane >> 4) * 8 + (e - 8)));
                bf[e] = (f16)(yb[(size_t)tn * D_EXP + d] * cb[d] * zb[(size_t)d * NPIX + tn]);
            }
            acc = wmma_(afrag[ks], bf, acc);
        }
        const int tok = n0 + (lane & 15);
#pragma unroll
        for (int r = 0; r < 8; ++r) {
            const int ch = m0 + r + (lane >> 4) * 8;
            out[(size_t)b * D_MODEL * NPIX + (size_t)ch * NPIX + tok] = acc[r] + bv[r];
        }
    }
}

// ============================== launcher ===================================
extern "C" void kernel_launch(void* const* d_in, const int* in_sizes, int n_in,
                              void* d_out, int out_size, void* d_ws, size_t ws_size,
                              hipStream_t stream) {
    const float* x        = (const float*)d_in[0];
    const float* in_w     = (const float*)d_in[1];
    const float* in_b     = (const float*)d_in[2];
    const float* dw_w     = (const float*)d_in[3];
    const float* dw_b     = (const float*)d_in[4];
    const float* x_proj_w = (const float*)d_in[5];
    const float* dt_w     = (const float*)d_in[6];
    const float* dt_b     = (const float*)d_in[7];
    const float* A_logs   = (const float*)d_in[8];
    const float* Ds       = (const float*)d_in[9];
    const float* onw      = (const float*)d_in[10];
    const float* onb      = (const float*)d_in[11];
    const float* cin_w    = (const float*)d_in[12];
    const float* cin_b    = (const float*)d_in[13];
    const float* cout_w   = (const float*)d_in[14];
    const float* cout_b   = (const float*)d_in[15];
    const float* xc_proj_w= (const float*)d_in[16];
    const float* dtc_w    = (const float*)d_in[17];
    const float* dtc_b    = (const float*)d_in[18];
    const float* Ac_logs  = (const float*)d_in[19];
    const float* Dsc      = (const float*)d_in[20];
    const float* cn_w     = (const float*)d_in[21];
    const float* cn_b     = (const float*)d_in[22];
    const float* out_w    = (const float*)d_in[23];
    const float* out_b    = (const float*)d_in[24];
    float* out = (float*)d_out;

    // scratch carve-up (~29.7 MiB), every buffer fully written before read
    float* ws   = (float*)d_ws;
    float* xz   = ws;                                   // 2*384*2304
    float* x1s  = xz   + (size_t)2 * 384 * NPIX;        // 2*2304*192
    float* proj = x1s  + (size_t)2 * NPIX * D_EXP;      // 2*2304*152
    float* y4   = proj + (size_t)2 * NPIX * NPROJ;      // 8*2304*192
    float* yln  = y4   + (size_t)8 * NPIX * D_EXP;      // 2*2304*192
    float* cbuf = yln  + (size_t)2 * NPIX * D_EXP;      // 2*192

    k_in_proj   <<<dim3(24, 9, 2), 32, 0, stream>>>(x, in_w, in_b, xz);
    k_dwconv    <<<dim3(3456), 256, 0, stream>>>(xz, dw_w, dw_b, x1s);
    k_proj      <<<dim3(48, 5, 2), 32, 0, stream>>>(x1s, x_proj_w, proj);
    k_scan      <<<dim3(48), 32, 0, stream>>>(x1s, proj, dt_w, dt_b, A_logs, Ds, y4);
    k_combine_ln<<<dim3(4608), 32, 0, stream>>>(y4, onw, onb, yln);
    k_channel   <<<dim3(2), 256, 0, stream>>>(yln, cin_w, cin_b, cout_w, cout_b,
                                              xc_proj_w, dtc_w, dtc_b, Ac_logs, Dsc,
                                              cn_w, cn_b, cbuf);
    k_out_proj  <<<dim3(6, 16, 2), 32, 0, stream>>>(yln, cbuf, xz, out_w, out_b, out);
}
